// WindowAttention_24584392802499
// MI455X (gfx1250) — compile-verified, matrix operand order checked by reading, not run on hardware
//
#include <hip/hip_runtime.h>
#include <hip/hip_bf16.h>

typedef __attribute__((ext_vector_type(16))) _Float16 v16h;
typedef __attribute__((ext_vector_type(8)))  _Float16 v8h;
typedef __attribute__((ext_vector_type(8)))  float    v8f;

#define N_TOK 49
#define NP    64          // padded tokens
#define CCH   128
#define HH    4
#define NWIN  1024

// LDS strides (elements). Row strides are multiples of 8 halves (16B) so
// v8h (b128) fragment loads are aligned; strides chosen to spread banks.
#define XS 136            // f16 stride for Xh / Qh / Kh / Oh  (272B = 17*16)
#define VS 72             // f16 stride for Vt and P           (144B = 9*16)
#define SS 66             // f32 stride for S and Mlds (scalar access only)

// LDS byte offsets (all multiples of 16)
#define OFF_X  0                        // 64*136*2  = 17408
#define OFF_Q  17408                    // 64*136*2  = 17408
#define OFF_K  34816                    // 64*136*2  = 17408
#define OFF_V  52224                    // 128*72*2  = 18432
#define OFF_S  70656                    // 4*64*66*4 = 67584
#define OFF_P  138240                   // 4*64*72*2 = 36864  (phase>=3)
#define OFF_M  138240                   // 64*66*4   = 16896  (phase<=2, overlaps P)
#define SMEM_BYTES 175104

// Workspace byte offsets
#define WSO_WQKVT 0                     // f16 [384][128]      = 98304
#define WSO_PROJT 98304                 // f16 [128][128]      = 32768
#define WSO_QKVB  131072                // f32 [384]           = 1536
#define WSO_BIAS  132608                // f32 [4][64][64] pad = 65536

#define WMMA_F16(A, B, C) __builtin_amdgcn_wmma_f32_16x16x32_f16( \
        false, (A), false, (B), (short)0, (C), false, false)

// ---------------------------------------------------------------------------
// Fragment loader: CDNA5 16-bit A/B layout (ISA 7.12.2).  For lane l,
// row = l&15, hi = l>>4; the 16 halves are K = hi*8..hi*8+7 (VGPR0..3) and
// K = 16+hi*8..16+hi*8+7 (VGPR4..7)  ->  exactly two aligned b128 loads.
// B operands are stored transposed (Bt[n][k]) so the same loader applies.
// ---------------------------------------------------------------------------
__device__ __forceinline__ v16h frag_ld(const _Float16* rowptr, int k0, int hi) {
    const v8h a = *(const v8h*)(rowptr + k0 + hi * 8);
    const v8h b = *(const v8h*)(rowptr + k0 + 16 + hi * 8);
    v16h r;
#pragma unroll
    for (int e = 0; e < 8; ++e) { r[e] = a[e]; r[e + 8] = b[e]; }
    return r;
}

// ---------------------------------------------------------------------------
// Prologue: f16 transposed/permuted weights, permuted+scaled qkv bias,
// zero-padded gathered relative-position bias [4][64][64].
// torch qkv reshape: out channel cold = (dd*H + h)*3 + s  maps to
// cnew = s*128 + h*32 + dd (q|k|v contiguous, head-major); 1/sqrt(d) folded
// into q columns and q bias.
// ---------------------------------------------------------------------------
__global__ __launch_bounds__(256) void prologue_kernel(
    const float* __restrict__ qkv_w, const float* __restrict__ qkv_b,
    const float* __restrict__ proj_w, const float* __restrict__ bias_table,
    const int* __restrict__ rel_idx,
    _Float16* __restrict__ wqkvt, _Float16* __restrict__ projt,
    float* __restrict__ qkvb, float* __restrict__ biasf)
{
    const float scale = 0.17677669529663687f;   // 32^-0.5
    int t = blockIdx.x * blockDim.x + threadIdx.x;
    if (t < 49152) {                                   // Wqkv^T (f16)
        int cnew = t >> 7, k = t & 127;
        int s = cnew >> 7, rem = cnew & 127;
        int h = rem >> 5, dd = rem & 31;
        int cold = (dd * HH + h) * 3 + s;
        float v = qkv_w[k * 384 + cold];
        if (s == 0) v *= scale;
        wqkvt[cnew * CCH + k] = (_Float16)v;
    } else if (t < 49536) {                            // qkv bias permuted
        int cnew = t - 49152;
        int s = cnew >> 7, rem = cnew & 127;
        int h = rem >> 5, dd = rem & 31;
        int cold = (dd * HH + h) * 3 + s;
        float v = qkv_b[cold];
        if (s == 0) v *= scale;
        qkvb[cnew] = v;
    } else if (t < 65920) {                            // proj^T (f16)
        int u = t - 49536;
        int c = u >> 7, k = u & 127;
        projt[c * CCH + k] = (_Float16)proj_w[k * CCH + c];
    } else if (t < 82304) {                            // padded rel-pos bias
        int u = t - 65920;                             // [4][64][64]
        int h = u >> 12, rem = u & 4095;
        int i = rem >> 6, j = rem & 63;
        float v = 0.f;
        if (i < N_TOK && j < N_TOK)
            v = bias_table[rel_idx[i * N_TOK + j] * HH + h];
        biasf[u] = v;
    }
}

// ---------------------------------------------------------------------------
// Main kernel: one workgroup (8 wave32) per window.  Each wave owns fixed
// M-rows per phase; A fragments hoisted into registers and reused across
// the N-tile sweep; all B fragments of a tile are loaded before the WMMA
// chain so they issue as one load clause.
// ---------------------------------------------------------------------------
__global__ __launch_bounds__(256) void win_attn_kernel(
    const float* __restrict__ x, const float* __restrict__ mask,
    const _Float16* __restrict__ wqkvt, const _Float16* __restrict__ projt,
    const float* __restrict__ qkvb, const float* __restrict__ biasf,
    const float* __restrict__ proj_b, float* __restrict__ out)
{
    extern __shared__ char smem[];
    _Float16* Xh   = (_Float16*)(smem + OFF_X);   // [64][136] tok x chan
    _Float16* Qh   = (_Float16*)(smem + OFF_Q);   // [64][136]
    _Float16* Kh   = (_Float16*)(smem + OFF_K);   // [64][136] (Bt for QK^T)
    _Float16* Vt   = (_Float16*)(smem + OFF_V);   // [128][72] V transposed
    float*    S    = (float*)   (smem + OFF_S);   // [4][64][66] logits
    _Float16* P    = (_Float16*)(smem + OFF_P);   // [4][64][72] probs
    float*    Mlds = (float*)   (smem + OFF_M);   // [64][66] padded mask
    _Float16* Oh   = Xh;                          // reuse Xh after phase 1

    const int b    = blockIdx.x;
    const int w    = b & (NWIN - 1);              // window index for mask
    const int t    = threadIdx.x;
    const int lane = t & 31;
    const int ln   = lane & 15;
    const int hi   = lane >> 4;
    // wave id as an SGPR: tile routing + loop bounds become scalar
    const int wv   = __builtin_amdgcn_readfirstlane(t) >> 5;

    // ---- phase 0: zero Xh + Mlds, then load/transpose X and stage mask ---
    {
        uint32_t* Z = (uint32_t*)Xh;
        for (int i = t; i < NP * XS / 2; i += 256) Z[i] = 0u;
        uint32_t* ZM = (uint32_t*)Mlds;
        for (int i = t; i < NP * SS; i += 256) ZM[i] = 0u;
    }
    __syncthreads();
    {
        const float* xb = x + (size_t)b * (CCH * N_TOK);
        for (int i = t; i < CCH * N_TOK; i += 256) {
            int c = i / N_TOK, n = i - c * N_TOK;
            Xh[n * XS + c] = (_Float16)xb[i];
        }
        const float* mw = mask + (size_t)w * N_TOK * N_TOK;
        for (int i = t; i < N_TOK * N_TOK; i += 256) {
            int r = i / N_TOK, c = i - r * N_TOK;
            Mlds[r * SS + c] = mw[i];
        }
    }
    __syncthreads();

    // ---- phase 1: QKV = X @ Wqkv^T + b ; wave owns mt, sweeps 12 N-tiles -
    {
        const int mt = wv >> 1, half = wv & 1;    // scalar
        const _Float16* arow = Xh + (mt * 16 + ln) * XS;
        const v16h a0 = frag_ld(arow, 0,  hi);
        const v16h a1 = frag_ld(arow, 32, hi);
        const v16h a2 = frag_ld(arow, 64, hi);
        const v16h a3 = frag_ld(arow, 96, hi);
        for (int j = 0; j < 12; ++j) {
            int nt = half * 12 + j;               // scalar
            int col = nt * 16 + ln;
            const _Float16* brow = wqkvt + (size_t)col * CCH;
            // load all B fragments first -> single 8x b128 load clause
            v16h b0 = frag_ld(brow, 0,  hi);
            v16h b1 = frag_ld(brow, 32, hi);
            v16h b2 = frag_ld(brow, 64, hi);
            v16h b3 = frag_ld(brow, 96, hi);
            float bval = qkvb[col];
            v8f acc;
#pragma unroll
            for (int r = 0; r < 8; ++r) acc[r] = bval;
            acc = WMMA_F16(a0, b0, acc);
            acc = WMMA_F16(a1, b1, acc);
            acc = WMMA_F16(a2, b2, acc);
            acc = WMMA_F16(a3, b3, acc);
            // routing depends only on scalar nt -> scalar branches
            if (nt < 8) {
#pragma unroll
                for (int r = 0; r < 8; ++r)
                    Qh[(mt * 16 + r + hi * 8) * XS + col] = (_Float16)acc[r];
            } else if (nt < 16) {
#pragma unroll
                for (int r = 0; r < 8; ++r)
                    Kh[(mt * 16 + r + hi * 8) * XS + (col - 128)] = (_Float16)acc[r];
            } else {
#pragma unroll
                for (int r = 0; r < 8; ++r)
                    Vt[(col - 256) * VS + (mt * 16 + r + hi * 8)] = (_Float16)acc[r];
            }
        }
    }
    __syncthreads();

    // ---- phase 2: S = Q K^T + bias + mask ; wave owns (h, mt pair) -------
    {
        const int h = wv >> 1, half = wv & 1;     // scalar
        float* Sh = S + h * NP * SS;
        const float* Bh = biasf + (h << 12);      // padded [64][64]
#pragma unroll
        for (int mm = 0; mm < 2; ++mm) {
            int mt = half * 2 + mm;
            const v16h a = frag_ld(Qh + (mt * 16 + ln) * XS + h * 32, 0, hi);
#pragma unroll
            for (int nt = 0; nt < 4; ++nt) {
                v16h bb = frag_ld(Kh + (nt * 16 + ln) * XS + h * 32, 0, hi);
                v8f acc = {};
                acc = WMMA_F16(a, bb, acc);
                int j = nt * 16 + ln;
#pragma unroll
                for (int r = 0; r < 8; ++r) {
                    int i2 = mt * 16 + r + hi * 8;
                    Sh[i2 * SS + j] = acc[r] + Bh[i2 * 64 + j] + Mlds[i2 * SS + j];
                }
            }
        }
    }
    __syncthreads();

    // ---- phase 3: softmax rows -> P (f16), pad rows/cols zeroed ----------
    {
        int h = t >> 6, i2 = t & 63;
        float*    Sr = S + (h * NP + i2) * SS;
        _Float16* Pr = P + (h * NP + i2) * VS;
        if (i2 < N_TOK) {
            float m = -1e30f;
            for (int j = 0; j < N_TOK; ++j) m = fmaxf(m, Sr[j]);
            float sum = 0.f;
            for (int j = 0; j < N_TOK; ++j) { float e = __expf(Sr[j] - m); Sr[j] = e; sum += e; }
            float inv = 1.f / sum;
            for (int j = 0; j < N_TOK; ++j) Pr[j] = (_Float16)(Sr[j] * inv);
            for (int j = N_TOK; j < NP; ++j) Pr[j] = (_Float16)0.f;
        } else {
            for (int j = 0; j < NP; ++j) Pr[j] = (_Float16)0.f;
        }
    }
    __syncthreads();

    // ---- phase 4: O = P @ V ; wave owns (h, mt pair), K-dim = 64 ---------
    {
        const int h = wv >> 1, half = wv & 1;     // scalar
#pragma unroll
        for (int mm = 0; mm < 2; ++mm) {
            int mt = half * 2 + mm;
            const _Float16* arow = P + (h * NP + mt * 16 + ln) * VS;
            const v16h a0 = frag_ld(arow, 0,  hi);
            const v16h a1 = frag_ld(arow, 32, hi);
#pragma unroll
            for (int nt = 0; nt < 2; ++nt) {
                int colc = h * 32 + nt * 16 + ln;
                const _Float16* brow = Vt + colc * VS;  // Vt is Bt for P@V
                v16h b0 = frag_ld(brow, 0,  hi);
                v16h b1 = frag_ld(brow, 32, hi);
                v8f acc = {};
                acc = WMMA_F16(a0, b0, acc);
                acc = WMMA_F16(a1, b1, acc);
#pragma unroll
                for (int r = 0; r < 8; ++r)
                    Oh[(mt * 16 + r + hi * 8) * XS + colc] = (_Float16)acc[r];
            }
        }
    }
    __syncthreads();

    // ---- phase 5: out = O @ proj^T + proj_b ; wave owns mt ---------------
    {
        const int mt = wv >> 1, half = wv & 1;    // scalar
        const _Float16* arow = Oh + (mt * 16 + ln) * XS;
        const v16h a0 = frag_ld(arow, 0,  hi);
        const v16h a1 = frag_ld(arow, 32, hi);
        const v16h a2 = frag_ld(arow, 64, hi);
        const v16h a3 = frag_ld(arow, 96, hi);
        float* ob = out + (size_t)b * (CCH * N_TOK);
        for (int j = 0; j < 4; ++j) {
            int nt = half * 4 + j;                // scalar
            int col = nt * 16 + ln;
            const _Float16* brow = projt + (size_t)col * CCH;
            v16h b0 = frag_ld(brow, 0,  hi);
            v16h b1 = frag_ld(brow, 32, hi);
            v16h b2 = frag_ld(brow, 64, hi);
            v16h b3 = frag_ld(brow, 96, hi);
            float bval = proj_b[col];
            v8f acc;
#pragma unroll
            for (int r = 0; r < 8; ++r) acc[r] = bval;
            acc = WMMA_F16(a0, b0, acc);
            acc = WMMA_F16(a1, b1, acc);
            acc = WMMA_F16(a2, b2, acc);
            acc = WMMA_F16(a3, b3, acc);
            if (mt < 3) {                         // rows mt*16..mt*16+15 < 49
#pragma unroll
                for (int r = 0; r < 8; ++r)
                    ob[col * N_TOK + (mt * 16 + r + hi * 8)] = acc[r];
            } else {                              // mt==3: only row 48 valid
                if (hi == 0) ob[col * N_TOK + 48] = acc[0];
            }
        }
    }
}

// ---------------------------------------------------------------------------
extern "C" void kernel_launch(void* const* d_in, const int* in_sizes, int n_in,
                              void* d_out, int out_size, void* d_ws, size_t ws_size,
                              hipStream_t stream) {
    const float* x          = (const float*)d_in[0];
    const float* qkv_w      = (const float*)d_in[1];
    const float* qkv_b      = (const float*)d_in[2];
    const float* proj_w     = (const float*)d_in[3];
    const float* proj_b     = (const float*)d_in[4];
    const float* bias_table = (const float*)d_in[5];
    const int*   rel_idx    = (const int*)d_in[6];
    const float* mask       = (const float*)d_in[7];
    float* out = (float*)d_out;

    char* ws = (char*)d_ws;
    _Float16* wqkvt = (_Float16*)(ws + WSO_WQKVT);
    _Float16* projt = (_Float16*)(ws + WSO_PROJT);
    float*    qkvb  = (float*)(ws + WSO_QKVB);
    float*    biasf = (float*)(ws + WSO_BIAS);

    prologue_kernel<<<322, 256, 0, stream>>>(qkv_w, qkv_b, proj_w, bias_table,
                                             rel_idx, wqkvt, projt, qkvb, biasf);
    win_attn_kernel<<<8192, 256, SMEM_BYTES, stream>>>(
        x, mask, wqkvt, projt, qkvb, biasf, proj_b, out);
}
